// basicLSTM_35957466202202
// MI455X (gfx1250) — matrix operand order repaired and checked by reading, not compiled, not run on hardware
//
#include <hip/hip_runtime.h>

typedef __attribute__((ext_vector_type(16))) _Float16 v16h;
typedef __attribute__((ext_vector_type(8)))  float    v8f;

#define LSTM_T 512
#define LSTM_H 10

static __device__ __forceinline__ v8f wmma_f16(v16h a, v16h b, v8f c) {
  // D = A(16x32 f16) x B(32x16 f16) + C(16x16 f32)
  return __builtin_amdgcn_wmma_f32_16x16x32_f16(false, a, false, b, (short)0, c, false, false);
}

static __device__ __forceinline__ float tanh_f(float x) {
#if __has_builtin(__builtin_amdgcn_tanhf)
  return __builtin_amdgcn_tanhf(x);
#else
  float e = __expf(2.0f * x);
  return 1.0f - 2.0f * __builtin_amdgcn_rcpf(e + 1.0f);
#endif
}
static __device__ __forceinline__ float sigm(float x) {
#if __has_builtin(__builtin_amdgcn_tanhf)
  return 0.5f + 0.5f * __builtin_amdgcn_tanhf(0.5f * x);
#else
  return __builtin_amdgcn_rcpf(1.0f + __expf(-x));
#endif
}

__global__ __launch_bounds__(128)
void lstm_wmma_kernel(const float* __restrict__ x,
                      const float* __restrict__ h0,
                      const float* __restrict__ c0,
                      const float* __restrict__ W_ih,
                      const float* __restrict__ W_hh,
                      const float* __restrict__ b_ih,
                      const float* __restrict__ b_hh,
                      const float* __restrict__ W1,
                      const float* __restrict__ b1p,
                      const float* __restrict__ W2,
                      const float* __restrict__ b2p,
                      float* __restrict__ out)
{
  const int  lane = threadIdx.x & 31;
  const int  wv   = threadIdx.x >> 5;
  const int  wave = blockIdx.x * 4 + wv;
  const int  base = wave * 32;          // 32 batch rows per wave (2 tiles of 16)
  const int  m    = lane & 15;
  const bool hi   = lane >= 16;

  __shared__ float xls[4][32 * 33];     // per-wave x chunk, transposed [t_local][m], stride 33
  float* xl = xls[wv];

  // ---------------- constant A matrices (weights), resident all 512 steps ----
  // A f16 16x32, contiguous K order used here (not AMD's table order): lanes 0-15 hold K=0..15 (elem e = K=e);
  //                     lanes 16-31 hold K=16..31 (all zero here).
  // K slots: 0..9 = W_hh row, 10 = W_ih (x), 11 = b_ih+b_hh (times constant 1).
  v16h Ag[4];
  #pragma unroll
  for (int g = 0; g < 4; ++g) {
    v16h a = {};
    const int row = g * 10 + m;
    if (m < 10) {
      if (!hi) {
        #pragma unroll
        for (int e = 0; e < 8; ++e) a[e] = (_Float16)W_hh[row * 10 + e];
        a[8] = (_Float16)W_hh[row * 10 + 8];
        a[9] = (_Float16)W_hh[row * 10 + 9];
        a[10] = (_Float16)W_ih[row];
        a[11] = (_Float16)(b_ih[row] + b_hh[row]);
      }
    }
    Ag[g] = a;
  }
  v16h A1 = {};                           // regressor layer 1: K10 slot = 0, K11 = b1
  {
    v16h a = {};
    if (m < 10) {
      if (!hi) {
        #pragma unroll
        for (int e = 0; e < 8; ++e) a[e] = (_Float16)W1[m * 10 + e];
        a[8] = (_Float16)W1[m * 10 + 8];
        a[9] = (_Float16)W1[m * 10 + 9];
        a[11] = (_Float16)b1p[m];
      }
    }
    A1 = a;
  }
  v16h A2 = {};                           // regressor layer 2 (row M=0 only), K10 = b2
  {
    v16h a = {};
    if (m == 0) {
      if (!hi) {
        #pragma unroll
        for (int e = 0; e < 8; ++e) a[e] = (_Float16)W2[e];
        a[8] = (_Float16)W2[8];
        a[9] = (_Float16)W2[9];
        a[10] = (_Float16)b2p[0];
      }
    }
    A2 = a;
  }

  // ---------------- state: c in D-layout (lane=batch col, slot r -> j=r+8*hi) --
  float cs0[8], cs1[8];
  #pragma unroll
  for (int r = 0; r < 8; ++r) {
    const int j = r + (hi ? 8 : 0);
    cs0[r] = (j < 10) ? c0[(base + m) * LSTM_H + j]      : 0.0f;
    cs1[r] = (j < 10) ? c0[(base + 16 + m) * LSTM_H + j] : 0.0f;
  }
  // h state lives as the B operand (f16). B 32x16, same contiguous K order as A: lanes 0-15 hold K=0..15
  // (elem e = K=e), lanes 16-31 hold K=16..31 (all zero here).
  v16h B0 = {}, B1 = {};
  if (!hi) {
    #pragma unroll
    for (int e = 0; e < 10; ++e) {
      B0[e] = (_Float16)h0[(base + m) * LSTM_H + e];
      B1[e] = (_Float16)h0[(base + 16 + m) * LSTM_H + e];
    }
    B0[11] = (_Float16)1.0f;
    B1[11] = (_Float16)1.0f;
  }

  for (int tc = 0; tc < LSTM_T; tc += 32) {
    // stage x chunk, transposed: xl[t_local*33 + m_local]; coalesced global reads
    for (int ml = 0; ml < 32; ++ml)
      xl[lane * 33 + ml] = x[(base + ml) * LSTM_T + tc + lane];

    #pragma unroll 1
    for (int tl = 0; tl < 32; ++tl) {
      const int   t  = tc + tl;
      const float xa = xl[tl * 33 + m];        // x_t for tile 0 (broadcast read)
      const float xb = xl[tl * 33 + 16 + m];   // x_t for tile 1

      // patch x into K-slot 10 of B (lanes 0-15 only)
      v16h b0 = B0, b1t = B1;
      b0[10]  = hi ? (_Float16)0.0f : (_Float16)xa;
      b1t[10] = hi ? (_Float16)0.0f : (_Float16)xb;

      const v8f z = {};
      // gates = [W_hh | W_ih | b] x [h ; x ; 1]   (4 WMMAs per tile)
      v8f di0 = wmma_f16(Ag[0], b0,  z);
      v8f df0 = wmma_f16(Ag[1], b0,  z);
      v8f dg0 = wmma_f16(Ag[2], b0,  z);
      v8f do0 = wmma_f16(Ag[3], b0,  z);
      v8f di1 = wmma_f16(Ag[0], b1t, z);
      v8f df1 = wmma_f16(Ag[1], b1t, z);
      v8f dg1 = wmma_f16(Ag[2], b1t, z);
      v8f do1 = wmma_f16(Ag[3], b1t, z);

      float hn0[8], hn1[8];
      #pragma unroll
      for (int r = 0; r < 8; ++r) {
        float cn0 = sigm(df0[r]) * cs0[r] + sigm(di0[r]) * tanh_f(dg0[r]);
        cs0[r] = cn0;
        hn0[r] = sigm(do0[r]) * tanh_f(cn0);
        float cn1 = sigm(df1[r]) * cs1[r] + sigm(di1[r]) * tanh_f(dg1[r]);
        cs1[r] = cn1;
        hn1[r] = sigm(do1[r]) * tanh_f(cn1);
      }

      // h[m][8], h[m][9] live in lane 16+m (D rows 8,9) -> pull down via xor-16
      const float h8_0 = __shfl_xor(hn0[0], 16, 32);
      const float h9_0 = __shfl_xor(hn0[1], 16, 32);
      const float h8_1 = __shfl_xor(hn1[0], 16, 32);
      const float h9_1 = __shfl_xor(hn1[1], 16, 32);

      v16h nb0 = {}, nb1 = {};
      if (!hi) {
        #pragma unroll
        for (int e = 0; e < 8; ++e) {
          nb0[e] = (_Float16)hn0[e];
          nb1[e] = (_Float16)hn1[e];
        }
        nb0[8] = (_Float16)h8_0; nb0[9] = (_Float16)h9_0; nb0[11] = (_Float16)1.0f;
        nb1[8] = (_Float16)h8_1; nb1[9] = (_Float16)h9_1; nb1[11] = (_Float16)1.0f;
      }
      B0 = nb0;
      B1 = nb1;

      // regressor: y1 = relu(W1*h + b1) ; D-layout of y1 feeds directly as B2
      v8f y10 = wmma_f16(A1, B0, z);
      v8f y11 = wmma_f16(A1, B1, z);
      float ry0[8], ry1[8];
      #pragma unroll
      for (int r = 0; r < 8; ++r) {
        ry0[r] = fmaxf(y10[r], 0.0f);
        ry1[r] = fmaxf(y11[r], 0.0f);
      }
      const float r8_0 = __shfl_xor(ry0[0], 16, 32);
      const float r9_0 = __shfl_xor(ry0[1], 16, 32);
      const float r8_1 = __shfl_xor(ry1[0], 16, 32);
      const float r9_1 = __shfl_xor(ry1[1], 16, 32);

      v16h q0 = {}, q1 = {};
      if (!hi) {
        #pragma unroll
        for (int e = 0; e < 8; ++e) {
          q0[e] = (_Float16)ry0[e];
          q1[e] = (_Float16)ry1[e];
        }
        q0[8] = (_Float16)r8_0; q0[9] = (_Float16)r9_0; q0[10] = (_Float16)1.0f;
        q1[8] = (_Float16)r8_1; q1[9] = (_Float16)r9_1; q1[10] = (_Float16)1.0f;
      }
      v8f yo0 = wmma_f16(A2, q0, z);   // row 0 = W2 . relu(..) + b2
      v8f yo1 = wmma_f16(A2, q1, z);

      if (!hi) {
        out[(base + m) * LSTM_T + t]      = yo0[0] + xa;
        out[(base + 16 + m) * LSTM_T + t] = yo1[0] + xb;
      }
    }
  }
}

extern "C" void kernel_launch(void* const* d_in, const int* in_sizes, int n_in,
                              void* d_out, int out_size, void* d_ws, size_t ws_size,
                              hipStream_t stream) {
  const float* x    = (const float*)d_in[0];
  const float* h0   = (const float*)d_in[1];
  const float* c0   = (const float*)d_in[2];
  const float* W_ih = (const float*)d_in[3];
  const float* W_hh = (const float*)d_in[4];
  const float* b_ih = (const float*)d_in[5];
  const float* b_hh = (const float*)d_in[6];
  const float* W1   = (const float*)d_in[7];
  const float* b1   = (const float*)d_in[8];
  const float* W2   = (const float*)d_in[9];
  const float* b2   = (const float*)d_in[10];
  float* out = (float*)d_out;

  dim3 grid(32);    // 32 blocks x 4 waves x 32 batch rows = 4096
  dim3 block(128);
  hipLaunchKernelGGL(lstm_wmma_kernel, grid, block, 0, stream,
                     x, h0, c0, W_ih, W_hh, b_ih, b_hh, W1, b1, W2, b2, out);
}
